// MGNN_40252433498177
// MI455X (gfx1250) — compile-verified, matrix operand order checked by reading, not run on hardware
//
#include <hip/hip_runtime.h>
#include <hip/hip_bf16.h>

// ---------------------------------------------------------------------------
// Types for WMMA
// ---------------------------------------------------------------------------
typedef __attribute__((ext_vector_type(16))) _Float16 v16h;
typedef __attribute__((ext_vector_type(8)))  _Float16 v8h;
typedef __attribute__((ext_vector_type(8)))  float    v8f;

#define GEMM_ACCUM 1
#define GEMM_RELU  2
#define GEMM_BIAS  4

#define TBM 128
#define TBN 128
#define TBK 32

// ---------------------------------------------------------------------------
// Tiled f16 WMMA GEMM:
//   D[M,N](f32) (+)= A[M,K](f16 row-major) x B[K,N](f16, given TRANSPOSED as
//   BT[N,K] row-major) with optional bias[N] / accumulate / ReLU epilogue.
// M%128 == N%128 == K%32 == 0 by construction (dims are 256/1024/2048/4096).
// 256 threads = 8 waves (4 in M x 2 in N); each wave computes a 32x64 strip
// = 2x4 WMMA tiles = 8 v_wmma per K-step. Register-prefetch pipeline over LDS.
// Both A and BT are K-contiguous, so every fragment is two ds_load_b128.
// BN=128 halves re-reads of the HBM-dominant adjacency (A) stream vs BN=64.
// ---------------------------------------------------------------------------
__global__ __launch_bounds__(256)
void gemm_f16_wmma(const _Float16* __restrict__ A,
                   const _Float16* __restrict__ BT,
                   float* __restrict__ D,
                   const float* __restrict__ bias,
                   int M, int N, int K, int flags)
{
    __shared__ _Float16 As[TBM][TBK];   // 8 KB
    __shared__ _Float16 Bs[TBN][TBK];   // 8 KB (transposed: [n][k])

    const int tid   = threadIdx.x;
    const int lane  = tid & 31;
    const int wave  = tid >> 5;         // 0..7
    const int waveM = wave >> 1;        // 0..3 (M direction, 32 rows each)
    const int waveN = wave & 1;         // 0..1 (N direction, 64 cols each)
    const int hlf   = lane >> 4;
    const int l16   = lane & 15;

    const int bm = blockIdx.x * TBM;
    const int bn = blockIdx.y * TBN;

    // staging coordinates: 4 lanes per row, 8 halves (16B) per lane
    const int sr = tid >> 2;            // 0..63
    const int sc = (tid & 3) * 8;       // 0,8,16,24

    v8f acc[2][4] = {};

    v8h aReg[2];   // A tile rows: sr, sr+64
    v8h bReg[2];   // BT tile rows: sr, sr+64

    auto loadGlobal = [&](int k0) {
        #pragma unroll
        for (int it = 0; it < 2; ++it)
            aReg[it] = *(const v8h*)(A + (size_t)(bm + sr + it * 64) * K + k0 + sc);
        #pragma unroll
        for (int it = 0; it < 2; ++it)
            bReg[it] = *(const v8h*)(BT + (size_t)(bn + sr + it * 64) * K + k0 + sc);
    };
    auto storeLDS = [&]() {
        #pragma unroll
        for (int it = 0; it < 2; ++it)
            *(v8h*)&As[sr + it * 64][sc] = aReg[it];
        #pragma unroll
        for (int it = 0; it < 2; ++it)
            *(v8h*)&Bs[sr + it * 64][sc] = bReg[it];
    };

    loadGlobal(0);

    for (int k0 = 0; k0 < K; k0 += TBK) {
        storeLDS();
        __syncthreads();

        if (k0 + TBK < K) loadGlobal(k0 + TBK);

        // fragments: per documented 16-bit A layout, each lane needs two
        // contiguous 8-half runs (16B each) -> 2x ds_load_b128 per fragment.
        v16h af[2], bf[4];
        #pragma unroll
        for (int s = 0; s < 2; ++s) {
            const int m = waveM * 32 + s * 16 + l16;
            const v8h lo = *(const v8h*)&As[m][hlf * 8];
            const v8h hi = *(const v8h*)&As[m][16 + hlf * 8];
            v16h a;
            #pragma unroll
            for (int i = 0; i < 8; ++i) { a[i] = lo[i]; a[8 + i] = hi[i]; }
            af[s] = a;
        }
        #pragma unroll
        for (int t = 0; t < 4; ++t) {
            const int n = waveN * 64 + t * 16 + l16;
            const v8h lo = *(const v8h*)&Bs[n][hlf * 8];
            const v8h hi = *(const v8h*)&Bs[n][16 + hlf * 8];
            v16h b;
            #pragma unroll
            for (int i = 0; i < 8; ++i) { b[i] = lo[i]; b[8 + i] = hi[i]; }
            bf[t] = b;
        }

        #pragma unroll
        for (int mi = 0; mi < 2; ++mi)
            #pragma unroll
            for (int ni = 0; ni < 4; ++ni)
                acc[mi][ni] = __builtin_amdgcn_wmma_f32_16x16x32_f16(
                    false, af[mi], false, bf[ni],
                    (short)0, acc[mi][ni], false, false);

        __syncthreads();
    }

    // epilogue per documented 32-bit C/D layout:
    // VGPR r holds row (r + hlf*8), column l16 of the 16x16 tile.
    #pragma unroll
    for (int mi = 0; mi < 2; ++mi) {
        #pragma unroll
        for (int ni = 0; ni < 4; ++ni) {
            const int col = bn + waveN * 64 + ni * 16 + l16;
            #pragma unroll
            for (int r = 0; r < 8; ++r) {
                const int row = bm + waveM * 32 + mi * 16 + r + hlf * 8;
                float v = acc[mi][ni][r];
                const size_t idx = (size_t)row * N + col;
                if (flags & GEMM_ACCUM) v += D[idx];
                if (flags & GEMM_BIAS)  v += bias[col];
                if (flags & GEMM_RELU)  v = fmaxf(v, 0.0f);
                D[idx] = v;
            }
        }
    }
}

// ---------------------------------------------------------------------------
// Row sums over last dim: A is [S*R, C] fp32, one block per row.
// ---------------------------------------------------------------------------
__global__ __launch_bounds__(256)
void rowsum_kernel(const float* __restrict__ A, float* __restrict__ sums, int C)
{
    __shared__ float red[256];
    const size_t row = blockIdx.x;
    const float* p = A + row * (size_t)C;
    float s = 0.0f;
    for (int c = threadIdx.x; c < C; c += 256) s += p[c];
    red[threadIdx.x] = s;
    __syncthreads();
    for (int off = 128; off > 0; off >>= 1) {
        if (threadIdx.x < off) red[threadIdx.x] += red[threadIdx.x + off];
        __syncthreads();
    }
    if (threadIdx.x == 0) sums[row] = red[0];
}

// ---------------------------------------------------------------------------
// Grouped row-normalized sum: out[r,c] = sum_{s in [s0,s0+ns)} A[s,r,c]/rowsum
// Output f16 (feeds WMMA A operands). Grid: (C/256, R).
// ---------------------------------------------------------------------------
__global__ __launch_bounds__(256)
void combine_kernel(const float* __restrict__ A, const float* __restrict__ sums,
                    _Float16* __restrict__ out, int s0, int ns, int R, int C)
{
    const int c = blockIdx.x * 256 + threadIdx.x;
    const int r = blockIdx.y;
    float acc = 0.0f;
    for (int s = 0; s < ns; ++s) {
        const float sv  = sums[(size_t)(s0 + s) * R + r];
        const float inv = (sv == 0.0f) ? 1.0f : 1.0f / sv;   // matches reference
        acc += A[((size_t)(s0 + s) * R + r) * C + c] * inv;
    }
    out[(size_t)r * C + c] = (_Float16)acc;
}

// ---------------------------------------------------------------------------
// fp32 -> f16 conversion (row-major, for WMMA A operands)
// ---------------------------------------------------------------------------
__global__ __launch_bounds__(256)
void cvt_kernel(const float* __restrict__ in, _Float16* __restrict__ out, size_t n)
{
    const size_t i = (size_t)blockIdx.x * 256 + threadIdx.x;
    if (i < n) out[i] = (_Float16)in[i];
}

// ---------------------------------------------------------------------------
// fp32 -> f16 convert + transpose: in[R][C] f32 -> out[C][R] f16.
// Produces K-contiguous BT operands for the GEMM. 32x32 LDS tile, 256 thr.
// R%32 == C%32 == 0 for all uses.
// ---------------------------------------------------------------------------
__global__ __launch_bounds__(256)
void cvt_t_kernel(const float* __restrict__ in, _Float16* __restrict__ out,
                  int R, int C)
{
    __shared__ _Float16 tile[32][33];
    const int c0 = blockIdx.x * 32;
    const int r0 = blockIdx.y * 32;
    const int tx = threadIdx.x & 31;
    const int ty = threadIdx.x >> 5;     // 0..7
    #pragma unroll
    for (int i = 0; i < 32; i += 8)
        tile[ty + i][tx] = (_Float16)in[(size_t)(r0 + ty + i) * C + c0 + tx];
    __syncthreads();
    #pragma unroll
    for (int i = 0; i < 32; i += 8)
        out[(size_t)(c0 + ty + i) * R + r0 + tx] = tile[tx][ty + i];
}

// ---------------------------------------------------------------------------
// Semantic attention over 3 channel views. One block (128 thr = ATT_H) per node.
// Emits fp32 (kept for reference math) and f16 row-major (A operand of fc GEMM).
// ---------------------------------------------------------------------------
__global__ __launch_bounds__(128)
void attention_kernel(const float* __restrict__ h0, const float* __restrict__ h1,
                      const float* __restrict__ h2,
                      const float* __restrict__ Wa1,  // [256,128]
                      const float* __restrict__ ba1,  // [128]
                      const float* __restrict__ Wa2,  // [128]
                      float* __restrict__ out, _Float16* __restrict__ out_h)
{
    __shared__ float zsh[3][256];
    __shared__ float red[128];
    __shared__ float w[3];
    __shared__ float alpha[3];

    const int node = blockIdx.x;
    const int t    = threadIdx.x;
    const float* hs[3] = { h0, h1, h2 };

    for (int j = 0; j < 3; ++j)
        for (int c = t; c < 256; c += 128)
            zsh[j][c] = hs[j][(size_t)node * 256 + c];
    __syncthreads();

    for (int j = 0; j < 3; ++j) {
        float a = ba1[t];
        #pragma unroll 4
        for (int c = 0; c < 256; ++c) a += zsh[j][c] * Wa1[c * 128 + t];
        a = fmaxf(a, 0.0f) * Wa2[t];
        red[t] = a;
        __syncthreads();
        for (int off = 64; off > 0; off >>= 1) {
            if (t < off) red[t] += red[t + off];
            __syncthreads();
        }
        if (t == 0) w[j] = red[0];
        __syncthreads();
    }

    if (t == 0) {
        const float m  = fmaxf(w[0], fmaxf(w[1], w[2]));
        const float e0 = __expf(w[0] - m), e1 = __expf(w[1] - m), e2 = __expf(w[2] - m);
        const float inv = 1.0f / (e0 + e1 + e2);
        alpha[0] = e0 * inv; alpha[1] = e1 * inv; alpha[2] = e2 * inv;
    }
    __syncthreads();

    for (int c = t; c < 256; c += 128) {
        const float v = alpha[0] * zsh[0][c] + alpha[1] * zsh[1][c] + alpha[2] * zsh[2][c];
        out[(size_t)node * 256 + c]   = v;
        out_h[(size_t)node * 256 + c] = (_Float16)v;
    }
}

// ---------------------------------------------------------------------------
// Host orchestration
// ---------------------------------------------------------------------------
extern "C" void kernel_launch(void* const* d_in, const int* in_sizes, int n_in,
                              void* d_out, int out_size, void* d_ws, size_t ws_size,
                              hipStream_t stream)
{
    (void)in_sizes; (void)n_in; (void)out_size; (void)ws_size;

    constexpr int ND  = 2048;
    constexpr int NP  = 4096;
    constexpr int DIM = 256;
    constexpr int DIN = 1024;

    const float* drug_feat    = (const float*)d_in[0];
    const float* protein_feat = (const float*)d_in[1];
    const float* A_dd = (const float*)d_in[2];   // [9,  ND, ND]
    const float* A_pd = (const float*)d_in[3];   // [10, ND, NP]
    const float* A_dp = (const float*)d_in[4];   // [10, NP, ND]
    const float* A_pp = (const float*)d_in[5];   // [9,  NP, NP]
    const float* Wproj_d = (const float*)d_in[6];
    const float* bproj_d = (const float*)d_in[7];
    const float* Wproj_p = (const float*)d_in[8];
    const float* bproj_p = (const float*)d_in[9];
    const float* Wdd = (const float*)d_in[10];
    const float* bdd = (const float*)d_in[11];
    const float* Wpd = (const float*)d_in[12];
    const float* bpd = (const float*)d_in[13];
    const float* Wdp = (const float*)d_in[14];
    const float* bdp = (const float*)d_in[15];
    const float* Wpp = (const float*)d_in[16];
    const float* bpp = (const float*)d_in[17];
    const float* Wfc_d = (const float*)d_in[18];
    const float* bfc_d = (const float*)d_in[19];
    const float* Wfc_p = (const float*)d_in[20];
    const float* bfc_p = (const float*)d_in[21];
    const float* Wa1 = (const float*)d_in[22];
    const float* ba1 = (const float*)d_in[23];
    const float* Wa2 = (const float*)d_in[24];
    // d_in[25] drug_drug, d_in[26] mask, d_in[27] n_layers: unused by output.

    // ---- workspace carve-up (256B aligned chunks) ----
    char* ws = (char*)d_ws;
    size_t off = 0;
    auto alloc = [&](size_t bytes) -> char* {
        char* p = ws + off;
        off += (bytes + 255) & ~(size_t)255;
        return p;
    };

    float* rs_dd = (float*)alloc(9ull  * ND * 4);
    float* rs_pd = (float*)alloc(10ull * ND * 4);
    float* rs_dp = (float*)alloc(10ull * NP * 4);
    float* rs_pp = (float*)alloc(9ull  * NP * 4);

    _Float16* Gdd = (_Float16*)alloc(3ull * ND * ND * 2);
    _Float16* Gpd = (_Float16*)alloc(3ull * ND * NP * 2);
    _Float16* Gdp = (_Float16*)alloc(3ull * NP * ND * 2);
    _Float16* Gpp = (_Float16*)alloc(3ull * NP * NP * 2);

    _Float16* df_h = (_Float16*)alloc((size_t)ND * DIN * 2);   // A operand
    _Float16* pf_h = (_Float16*)alloc((size_t)NP * DIN * 2);   // A operand

    // transposed f16 weights (BT operands, [N][K])
    _Float16* wprojdT = (_Float16*)alloc((size_t)DIM * DIN * 2);
    _Float16* wprojpT = (_Float16*)alloc((size_t)DIM * DIN * 2);
    _Float16* wddT = (_Float16*)alloc(3ull * DIM * DIM * 2);
    _Float16* wpdT = (_Float16*)alloc(3ull * DIM * DIM * 2);
    _Float16* wdpT = (_Float16*)alloc(3ull * DIM * DIM * 2);
    _Float16* wppT = (_Float16*)alloc(3ull * DIM * DIM * 2);
    _Float16* wfcdT = (_Float16*)alloc((size_t)DIM * DIM * 2);
    _Float16* wfcpT = (_Float16*)alloc((size_t)DIM * DIM * 2);

    float*    dbuf = (float*)alloc((size_t)ND * DIM * 4);
    float*    pbuf = (float*)alloc((size_t)NP * DIM * 4);
    _Float16* d_h  = (_Float16*)alloc((size_t)ND * DIM * 2);   // A operand
    _Float16* p_h  = (_Float16*)alloc((size_t)NP * DIM * 2);   // A operand

    float*    tmp32 = (float*)alloc((size_t)NP * DIM * 4);

    // transposed activations (BT operands of the big propagation GEMMs):
    // pdT[k]  : [DIM][ND], ppdT[k]/pppT[k] : [DIM][NP], pdpT[k] : [DIM][ND]
    _Float16* pdT  = (_Float16*)alloc(3ull * DIM * ND * 2);
    _Float16* ppdT = (_Float16*)alloc(3ull * DIM * NP * 2);
    _Float16* pdpT = (_Float16*)alloc(3ull * DIM * ND * 2);
    _Float16* pppT = (_Float16*)alloc(3ull * DIM * NP * 2);

    float* hd = (float*)alloc(3ull * ND * DIM * 4);   // [j][ND][DIM], j: 0=h0,1=h1,2=h2
    float* hp = (float*)alloc(3ull * NP * DIM * 4);

    float*    attd   = (float*)alloc((size_t)ND * DIM * 4);
    float*    attp   = (float*)alloc((size_t)NP * DIM * 4);
    _Float16* attd_h = (_Float16*)alloc((size_t)ND * DIM * 2); // A operand
    _Float16* attp_h = (_Float16*)alloc((size_t)NP * DIM * 2); // A operand

    // ---- helpers ----
    auto gemm = [&](const _Float16* A, const _Float16* BT, float* D,
                    const float* bias, int M, int N, int K, int flags) {
        dim3 grid(M / TBM, N / TBN);
        gemm_f16_wmma<<<grid, 256, 0, stream>>>(A, BT, D, bias, M, N, K, flags);
    };
    auto cvt = [&](const float* in, _Float16* out, size_t n) {
        cvt_kernel<<<(unsigned)((n + 255) / 256), 256, 0, stream>>>(in, out, n);
    };
    auto cvtT = [&](const float* in, _Float16* out, int R, int C) {
        dim3 grid(C / 32, R / 32);
        cvt_t_kernel<<<grid, 256, 0, stream>>>(in, out, R, C);
    };
    auto combine = [&](const float* A, const float* rs, _Float16* out,
                       int s0, int ns, int R, int C) {
        dim3 grid(C / 256, R);
        combine_kernel<<<grid, 256, 0, stream>>>(A, rs, out, s0, ns, R, C);
    };

    // ---- 1) adjacency row sums ----
    rowsum_kernel<<<9  * ND, 256, 0, stream>>>(A_dd, rs_dd, ND);
    rowsum_kernel<<<10 * ND, 256, 0, stream>>>(A_pd, rs_pd, NP);
    rowsum_kernel<<<10 * NP, 256, 0, stream>>>(A_dp, rs_dp, ND);
    rowsum_kernel<<<9  * NP, 256, 0, stream>>>(A_pp, rs_pp, NP);

    // ---- 2) grouped normalized adjacencies (f16), "use order" [0]=g1,[1]=g2,[2]=g0 ----
    combine(A_dd, rs_dd, Gdd + 0ull * ND * ND, 0, 5, ND, ND);
    combine(A_dd, rs_dd, Gdd + 1ull * ND * ND, 5, 3, ND, ND);
    combine(A_dd, rs_dd, Gdd + 2ull * ND * ND, 8, 1, ND, ND);
    combine(A_pd, rs_pd, Gpd + 0ull * ND * NP, 0, 4, ND, NP);
    combine(A_pd, rs_pd, Gpd + 1ull * ND * NP, 4, 2, ND, NP);
    combine(A_pd, rs_pd, Gpd + 2ull * ND * NP, 6, 4, ND, NP);
    combine(A_dp, rs_dp, Gdp + 0ull * NP * ND, 0, 4, NP, ND);
    combine(A_dp, rs_dp, Gdp + 1ull * NP * ND, 4, 2, NP, ND);
    combine(A_dp, rs_dp, Gdp + 2ull * NP * ND, 6, 4, NP, ND);
    combine(A_pp, rs_pp, Gpp + 0ull * NP * NP, 0, 5, NP, NP);
    combine(A_pp, rs_pp, Gpp + 1ull * NP * NP, 5, 3, NP, NP);
    combine(A_pp, rs_pp, Gpp + 2ull * NP * NP, 8, 1, NP, NP);

    // ---- 3) f16 copies of features (A operands) and transposed weights (BT) ----
    cvt(drug_feat,    df_h, (size_t)ND * DIN);
    cvt(protein_feat, pf_h, (size_t)NP * DIN);
    cvtT(Wproj_d, wprojdT, DIN, DIM);
    cvtT(Wproj_p, wprojpT, DIN, DIM);
    for (int k = 0; k < 3; ++k) {
        cvtT(Wdd + (size_t)k * DIM * DIM, wddT + (size_t)k * DIM * DIM, DIM, DIM);
        cvtT(Wpd + (size_t)k * DIM * DIM, wpdT + (size_t)k * DIM * DIM, DIM, DIM);
        cvtT(Wdp + (size_t)k * DIM * DIM, wdpT + (size_t)k * DIM * DIM, DIM, DIM);
        cvtT(Wpp + (size_t)k * DIM * DIM, wppT + (size_t)k * DIM * DIM, DIM, DIM);
    }
    cvtT(Wfc_d, wfcdT, DIM, DIM);
    cvtT(Wfc_p, wfcpT, DIM, DIM);

    // ---- 4) input projections ----
    gemm(df_h, wprojdT, dbuf, bproj_d, ND, DIM, DIN, GEMM_BIAS);
    cvt(dbuf, d_h, (size_t)ND * DIM);
    gemm(pf_h, wprojpT, pbuf, bproj_p, NP, DIM, DIN, GEMM_BIAS);
    cvt(pbuf, p_h, (size_t)NP * DIM);

    // ---- 5) GNN layers (n_layers == 2 in the reference) ----
    const size_t dC = (size_t)ND * DIM;
    const size_t pC = (size_t)NP * DIM;
    for (int layer = 0; layer < 2; ++layer) {
        // per-relation projections (f32 out -> transposed f16 BT operands)
        for (int k = 0; k < 3; ++k) {
            gemm(d_h, wddT + (size_t)k * DIM * DIM, tmp32, bdd + k * DIM,
                 ND, DIM, DIM, GEMM_BIAS);
            cvtT(tmp32, pdT + k * dC, ND, DIM);
            gemm(p_h, wpdT + (size_t)k * DIM * DIM, tmp32, bpd + k * DIM,
                 NP, DIM, DIM, GEMM_BIAS);
            cvtT(tmp32, ppdT + k * pC, NP, DIM);
            gemm(d_h, wdpT + (size_t)k * DIM * DIM, tmp32, bdp + k * DIM,
                 ND, DIM, DIM, GEMM_BIAS);
            cvtT(tmp32, pdpT + k * dC, ND, DIM);
            gemm(p_h, wppT + (size_t)k * DIM * DIM, tmp32, bpp + k * DIM,
                 NP, DIM, DIM, GEMM_BIAS);
            cvtT(tmp32, pppT + k * pC, NP, DIM);
        }

        // drug side: h1 (group 1), h2 (group 2), h0 (group 0)
        gemm(Gdd + 0ull * ND * ND, pdT  + 0 * dC, hd + 1 * dC, nullptr, ND, DIM, ND, 0);
        gemm(Gpd + 0ull * ND * NP, ppdT + 0 * pC, hd + 1 * dC, nullptr, ND, DIM, NP, GEMM_ACCUM | GEMM_RELU);
        gemm(Gdd + 1ull * ND * ND, pdT  + 1 * dC, hd + 2 * dC, nullptr, ND, DIM, ND, 0);
        gemm(Gpd + 1ull * ND * NP, ppdT + 1 * pC, hd + 2 * dC, nullptr, ND, DIM, NP, GEMM_ACCUM | GEMM_RELU);
        gemm(Gdd + 2ull * ND * ND, pdT  + 2 * dC, hd + 0 * dC, nullptr, ND, DIM, ND, 0);
        gemm(Gpd + 2ull * ND * NP, ppdT + 2 * pC, hd + 0 * dC, nullptr, ND, DIM, NP, GEMM_ACCUM | GEMM_RELU);

        // protein side
        gemm(Gdp + 0ull * NP * ND, pdpT + 0 * dC, hp + 1 * pC, nullptr, NP, DIM, ND, 0);
        gemm(Gpp + 0ull * NP * NP, pppT + 0 * pC, hp + 1 * pC, nullptr, NP, DIM, NP, GEMM_ACCUM | GEMM_RELU);
        gemm(Gdp + 1ull * NP * ND, pdpT + 1 * dC, hp + 2 * pC, nullptr, NP, DIM, ND, 0);
        gemm(Gpp + 1ull * NP * NP, pppT + 1 * pC, hp + 2 * pC, nullptr, NP, DIM, NP, GEMM_ACCUM | GEMM_RELU);
        gemm(Gpp + 2ull * NP * NP, pppT + 2 * pC, hp + 0 * pC, nullptr, NP, DIM, NP, 0);
        gemm(Gdp + 2ull * NP * ND, pdpT + 2 * dC, hp + 0 * pC, nullptr, NP, DIM, ND, GEMM_ACCUM | GEMM_RELU);

        // semantic attention (channel order: h0, h1, h2)
        attention_kernel<<<ND, 128, 0, stream>>>(hd + 0 * dC, hd + 1 * dC, hd + 2 * dC,
                                                 Wa1, ba1, Wa2, attd, attd_h);
        attention_kernel<<<NP, 128, 0, stream>>>(hp + 0 * pC, hp + 1 * pC, hp + 2 * pC,
                                                 Wa1, ba1, Wa2, attp, attp_h);

        // fc + relu -> new embeddings
        gemm(attd_h, wfcdT, dbuf, bfc_d, ND, DIM, DIM, GEMM_BIAS | GEMM_RELU);
        cvt(dbuf, d_h, dC);
        gemm(attp_h, wfcpT, pbuf, bfc_p, NP, DIM, DIM, GEMM_BIAS | GEMM_RELU);
        cvt(pbuf, p_h, pC);
    }

    // ---- 6) concat [d; p] into d_out (fp32) ----
    hipMemcpyAsync(d_out, dbuf, dC * sizeof(float), hipMemcpyDeviceToDevice, stream);
    hipMemcpyAsync((float*)d_out + dC, pbuf, pC * sizeof(float),
                   hipMemcpyDeviceToDevice, stream);
}
